// LinearDeepseekV3MoE_9990093931257
// MI455X (gfx1250) — matrix operand order repaired and checked by reading, not compiled, not run on hardware
//
#include <hip/hip_runtime.h>

// ---------------- model constants (DeepseekV3 MoE config from reference) ----
#define E_     32       // routed experts
#define H_     1024     // hidden
#define I_RT   512      // routed intermediate
#define G_     8        // groups
#define TKG_   4        // topk groups
#define TOPK_  8        // experts per token
#define SCALE_ 2.5f     // routed_scaling_factor

// ---------------- vector types (clang ext-vectors: trivially constructible) -
typedef __attribute__((ext_vector_type(16))) __bf16        v16bf;
typedef __attribute__((ext_vector_type(8)))  float         v8f;
typedef __attribute__((ext_vector_type(4)))  float         f32x4;
typedef __attribute__((ext_vector_type(4)))  unsigned int  u32x4;
typedef __attribute__((ext_vector_type(2)))  unsigned int  u32x2;

// fp32 -> bf16 scalar, round-to-nearest-even
__device__ __forceinline__ unsigned short f2bf(float f) {
  unsigned u = __builtin_bit_cast(unsigned, f);
  u += 0x7fffu + ((u >> 16) & 1u);
  return (unsigned short)(u >> 16);
}

// packed 2 x fp32 -> 2 x bf16 in one dword (hardware cvt when available)
#if __has_builtin(__builtin_amdgcn_cvt_pk_bf16_f32)
typedef __attribute__((ext_vector_type(2))) __bf16 v2bf;
__device__ __forceinline__ unsigned pk_bf16(float a, float b) {
  v2bf r = __builtin_amdgcn_cvt_pk_bf16_f32(a, b);
  return __builtin_bit_cast(unsigned, r);
}
#else
__device__ __forceinline__ unsigned pk_bf16(float a, float b) {
  return (unsigned)f2bf(a) | ((unsigned)f2bf(b) << 16);
}
#endif

union ABFrag {
  u32x4          q[2];
  unsigned int   u32[8];
  unsigned short s[16];
  v16bf          v;
};

// B fragment (32x16, bf16): lane holds col = lane&15, K = (lane>>4)*16 + j,
// j = 0..15 contiguous.
// fp32 source: 16 consecutive floats per lane, packed-cvt in regs.
__device__ __forceinline__ v16bf load_b_f32(const float* __restrict__ p) {
  ABFrag u;
  const f32x4* q = (const f32x4*)p;   // 64B-aligned (offsets are x16 floats)
#pragma unroll
  for (int i = 0; i < 4; ++i) {
    f32x4 f = q[i];
    u.u32[i * 2 + 0] = pk_bf16(f.x, f.y);
    u.u32[i * 2 + 1] = pk_bf16(f.z, f.w);
  }
  return u.v;
}

template <bool BF16W>
__device__ __forceinline__ v16bf load_b(const void* __restrict__ W, size_t off) {
  if constexpr (BF16W) {   // pre-converted bf16 weights: straight 2x b128 loads
    const unsigned short* p = (const unsigned short*)W + off;
    ABFrag u;
    u.q[0] = *(const u32x4*)p;
    u.q[1] = *(const u32x4*)(p + 8);
    return u.v;
  } else {
    return load_b_f32((const float*)W + off);
  }
}

// A fragment (16x32, bf16) from LDS row `base`: halves 0..15 contiguous
// at K=koff..koff+15.
__device__ __forceinline__ v16bf load_a_frag(const unsigned short* base, int koff) {
  ABFrag u;
  u.q[0] = *(const u32x4*)(base + koff);
  u.q[1] = *(const u32x4*)(base + koff + 8);
  return u.v;
}

__device__ __forceinline__ v8f wmma_bf16(v16bf a, v16bf b, v8f c) {
  // (neg_a, A, neg_b, B, c_mod, C, reuse_a, reuse_b)
  return __builtin_amdgcn_wmma_f32_16x16x32_bf16(false, a, false, b, (short)0, c,
                                                 false, false);
}

// ---------------- init: zero output accumulator + expert counters ----------
__global__ void init_kernel(float* __restrict__ out, size_t n,
                            int* __restrict__ counts) {
  size_t i = (size_t)blockIdx.x * blockDim.x + threadIdx.x;
  if (i < n) out[i] = 0.0f;
  if (i < E_) counts[i] = 0;
}

// ---------------- fp32 -> bf16 weight pre-conversion -----------------------
__global__ void cvt_bf16_kernel(const float* __restrict__ src,
                                unsigned short* __restrict__ dst, size_t n) {
  size_t i = ((size_t)blockIdx.x * blockDim.x + threadIdx.x) * 8;
  if (i + 8 <= n) {
    f32x4 a = *(const f32x4*)(src + i);
    f32x4 b = *(const f32x4*)(src + i + 4);
    u32x4 q;
    q.x = pk_bf16(a.x, a.y);
    q.y = pk_bf16(a.z, a.w);
    q.z = pk_bf16(b.x, b.y);
    q.w = pk_bf16(b.z, b.w);
    *(u32x4*)(dst + i) = q;
  } else {
    for (; i < n; ++i) dst[i] = f2bf(src[i]);
  }
}

// ---------------- router: exact fp32, one wave per token -------------------
__global__ void __launch_bounds__(256) router_kernel(
    const float* __restrict__ h, const float* __restrict__ gw,
    const float* __restrict__ gb, int* __restrict__ counts,
    int* __restrict__ tok_list, float* __restrict__ wgt_list, int T) {
  __shared__ float s_scores[8][E_];
  __shared__ float s_sc[8][E_];
  const int wave = threadIdx.x >> 5, lane = threadIdx.x & 31;
  const int t = blockIdx.x * 8 + wave;

  if (t < T) {  // lane e computes logit for expert e (E_ == wave32 lanes)
    const f32x4* hv = (const f32x4*)(h + (size_t)t * H_);
    const f32x4* wv = (const f32x4*)(gw + (size_t)lane * H_);
    float acc = 0.f;
#pragma unroll 4
    for (int k = 0; k < H_ / 4; ++k) {
      f32x4 a = hv[k], b = wv[k];
      acc += a.x * b.x + a.y * b.y + a.z * b.z + a.w * b.w;
    }
    float sg = 1.f / (1.f + __expf(-acc));
    s_scores[wave][lane] = sg;
    s_sc[wave][lane] = sg + gb[lane];
  }
  __syncthreads();

  if (t < T && lane == 0) {  // tiny serial grouped top-k (exact tie-break)
    float sc[E_], scores[E_];
    for (int i = 0; i < E_; ++i) { sc[i] = s_sc[wave][i]; scores[i] = s_scores[wave][i]; }
    float gsc[G_];
    for (int g = 0; g < G_; ++g) {  // sum of top-2 within group of 4
      const float* v = &sc[g * 4];
      int im = 0;
      for (int i = 1; i < 4; ++i) if (v[i] > v[im]) im = i;
      float s2 = -1e30f;
      for (int i = 0; i < 4; ++i) if (i != im && v[i] > s2) s2 = v[i];
      gsc[g] = v[im] + s2;
    }
    bool gmask[G_] = {};
    for (int k = 0; k < TKG_; ++k) {
      int best = 0; float bv = -1e30f;
      for (int g = 0; g < G_; ++g)
        if (!gmask[g] && gsc[g] > bv) { bv = gsc[g]; best = g; }
      gmask[best] = true;
    }
    float masked[E_];
    for (int i = 0; i < E_; ++i) masked[i] = gmask[i >> 2] ? sc[i] : 0.f;
    int idx[TOPK_]; float tw[TOPK_]; float wsum = 0.f;
    for (int k = 0; k < TOPK_; ++k) {
      int best = 0; float bv = -1e30f;
      for (int i = 0; i < E_; ++i)
        if (masked[i] > bv) { bv = masked[i]; best = i; }
      masked[best] = -1e30f;
      idx[k] = best; tw[k] = scores[best]; wsum += tw[k];
    }
    const float inv = SCALE_ / (wsum + 1e-20f);
    for (int k = 0; k < TOPK_; ++k) {
      int e = idx[k];
      int pos = atomicAdd(&counts[e], 1);
      tok_list[(size_t)e * T + pos] = t;
      wgt_list[(size_t)e * T + pos] = tw[k] * inv;
    }
  }
}

// ---------------- fused expert FFN: gate/up -> silu*mul -> down ------------
// One workgroup = 8 waves = one (expert, 32-token) tile.
// tok_list==nullptr  => identity token list with weight 1 (shared experts).
// BF16W: weights already converted to bf16 (workspace path).
template <int IDIM, bool BF16W>
__global__ void __launch_bounds__(256) expert_ffn_kernel(
    const float* __restrict__ h,
    const void* __restrict__ Wg,   // [slots, IDIM, H]
    const void* __restrict__ Wu,   // [slots, IDIM, H]
    const void* __restrict__ Wd,   // [slots, H, IDIM]
    const int* __restrict__ tok_list, const float* __restrict__ wgt_list,
    const int* __restrict__ counts, float* __restrict__ out, int T) {
  static_assert(IDIM % 512 == 0, "IDIM must be a multiple of 512");
  constexpr int HP = H_ + 8;     // LDS pitch (halves), bank-skewed
  constexpr int AP = IDIM + 8;
  __shared__ unsigned short sh_h[32 * HP];
  __shared__ unsigned short sh_act[32 * AP];
  __shared__ int   s_tok[32];
  __shared__ float s_wgt[32];

  const int e   = blockIdx.y;
  const int cnt = counts ? counts[e] : T;
  const int m0  = blockIdx.x * 32;
  if (m0 >= cnt) return;                       // uniform exit, EXEC stays full
  const int tid = threadIdx.x;

  if (tid < 32) {
    int m = m0 + tid;
    if (m < cnt) {
      s_tok[tid] = tok_list ? tok_list[(size_t)e * T + m] : m;
      s_wgt[tid] = wgt_list ? wgt_list[(size_t)e * T + m] : 1.0f;
    } else { s_tok[tid] = -1; s_wgt[tid] = 0.f; }
  }
  __syncthreads();

  // ---- stage gathered hidden tile 32 x 1024, fp32 -> bf16 LDS ----
  {
    const int r  = tid >> 3;            // 32 rows, 8 threads per row
    const int c0 = (tid & 7) * 128;
    const int tok = s_tok[r];
    const float* hrow = h + (size_t)(tok < 0 ? 0 : tok) * H_;
#pragma unroll 4
    for (int c = c0; c < c0 + 128; c += 4) {
      f32x4 v = {0.f, 0.f, 0.f, 0.f};
      if (tok >= 0) v = *(const f32x4*)(hrow + c);
      u32x2 p;
      p.x = pk_bf16(v.x, v.y);
      p.y = pk_bf16(v.z, v.w);
      *(u32x2*)&sh_h[r * HP + c] = p;
    }
  }
  __syncthreads();

  const int wave  = tid >> 5, lane = tid & 31;
  const int laneN = lane & 15;
  const int laneK = lane >> 4;        // 0 | 1
  const int abase = laneK * 16;       // A-frag K sub-offset
  const int kb    = laneK * 16;       // B-frag K sub-offset

  // ---- phase 1: gate & up projections (I columns, 512 per pass) ----
  for (int pass = 0; pass < IDIM / 512; ++pass) {
    const int ncw = pass * 512 + wave * 64;   // this wave's 64 I-columns
    v8f accg[2][4] = {}, accu[2][4] = {};
    size_t rowoff[4];
#pragma unroll
    for (int nt = 0; nt < 4; ++nt)
      rowoff[nt] = ((size_t)e * IDIM + ncw + nt * 16 + laneN) * H_ + kb;

    for (int k0 = 0; k0 < H_; k0 += 32) {
      v16bf a[2];
#pragma unroll
      for (int mt = 0; mt < 2; ++mt)
        a[mt] = load_a_frag(&sh_h[(mt * 16 + laneN) * HP], k0 + abase);
#pragma unroll
      for (int nt = 0; nt < 4; ++nt) {
        v16bf bg = load_b<BF16W>(Wg, rowoff[nt] + k0);
        v16bf bu = load_b<BF16W>(Wu, rowoff[nt] + k0);
#pragma unroll
        for (int mt = 0; mt < 2; ++mt) {
          accg[mt][nt] = wmma_bf16(a[mt], bg, accg[mt][nt]);
          accu[mt][nt] = wmma_bf16(a[mt], bu, accu[mt][nt]);
        }
      }
    }
    // silu(g) * u  -> bf16 activation tile in LDS
#pragma unroll
    for (int mt = 0; mt < 2; ++mt)
#pragma unroll
      for (int nt = 0; nt < 4; ++nt)
#pragma unroll
        for (int r = 0; r < 8; ++r) {
          float g = accg[mt][nt][r], u = accu[mt][nt][r];
          float act = g * (1.f / (1.f + __expf(-g))) * u;
          int m = mt * 16 + laneK * 8 + r;          // C layout: M = r + (l>=16)*8
          sh_act[m * AP + ncw + nt * 16 + laneN] = f2bf(act);
        }
  }
  __syncthreads();

  // ---- phase 2: down projection (H output columns, 128 per wave) ----
  {
    const int ncw = wave * 128;
    v8f accd[2][8] = {};
    size_t rowoff[8];
#pragma unroll
    for (int nt = 0; nt < 8; ++nt)
      rowoff[nt] = ((size_t)e * H_ + ncw + nt * 16 + laneN) * IDIM + kb;

    for (int k0 = 0; k0 < IDIM; k0 += 32) {
      v16bf a[2];
#pragma unroll
      for (int mt = 0; mt < 2; ++mt)
        a[mt] = load_a_frag(&sh_act[(mt * 16 + laneN) * AP], k0 + abase);
#pragma unroll
      for (int nt = 0; nt < 8; ++nt) {
        v16bf b = load_b<BF16W>(Wd, rowoff[nt] + k0);
#pragma unroll
        for (int mt = 0; mt < 2; ++mt)
          accd[mt][nt] = wmma_bf16(a[mt], b, accd[mt][nt]);
      }
    }
    // combine-weight scale + accumulate into output (fp32 global atomics)
#pragma unroll
    for (int mt = 0; mt < 2; ++mt)
#pragma unroll
      for (int r = 0; r < 8; ++r) {
        int m = mt * 16 + laneK * 8 + r;
        int tok = s_tok[m];
        if (tok < 0) continue;
        float w = s_wgt[m];
        float* orow = out + (size_t)tok * H_ + ncw + laneN;
#pragma unroll
        for (int nt = 0; nt < 8; ++nt)
          unsafeAtomicAdd(orow + nt * 16, accd[mt][nt][r] * w);
      }
  }
}

// ---------------- host-side launcher ---------------------------------------
extern "C" void kernel_launch(void* const* d_in, const int* in_sizes, int n_in,
                              void* d_out, int out_size, void* d_ws, size_t ws_size,
                              hipStream_t stream) {
  const float* h      = (const float*)d_in[0];
  const float* gate_w = (const float*)d_in[1];
  const float* gate_b = (const float*)d_in[2];
  float* out = (float*)d_out;
  const int T = in_sizes[0] / H_;          // 2048

  // workspace: counters + per-expert (token, weight) lists  (~2.1 MB),
  // then (optionally) bf16 copies of the 6 weight tensors (~106 MB).
  char* ws = (char*)d_ws;
  int* counts   = (int*)ws;  ws += 256;
  int* tok_list = (int*)ws;  ws += (size_t)E_ * T * sizeof(int);
  float* wgt_list = (float*)ws;  ws += (size_t)E_ * T * sizeof(float);

  size_t wn[6];
  size_t need = 256 + 2ull * E_ * T * 4;
  for (int i = 0; i < 6; ++i) { wn[i] = (size_t)in_sizes[3 + i]; need += wn[i] * 2; }
  const bool pre = ws_size >= need;        // bf16 weight path fits in scratch?

  unsigned short* wbf[6] = {};
  if (pre) {
    char* p = ws;
    for (int i = 0; i < 6; ++i) {
      wbf[i] = (unsigned short*)p;
      p += wn[i] * 2;
      cvt_bf16_kernel<<<(unsigned)((wn[i] / 8 + 255) / 256), 256, 0, stream>>>(
          (const float*)d_in[3 + i], wbf[i], wn[i]);
    }
  }

  const size_t n = (size_t)T * H_;
  init_kernel<<<(unsigned)((n + 255) / 256), 256, 0, stream>>>(out, n, counts);
  router_kernel<<<(T + 7) / 8, 256, 0, stream>>>(h, gate_w, gate_b, counts,
                                                 tok_list, wgt_list, T);
  dim3 grid_r((T + 31) / 32, E_);          // most tiles exit early (count test)
  dim3 grid_s((T + 31) / 32, 1);           // shared experts: IDIM = 1024
  if (pre) {
    expert_ffn_kernel<I_RT, true><<<grid_r, 256, 0, stream>>>(
        h, wbf[0], wbf[1], wbf[2], tok_list, wgt_list, counts, out, T);
    expert_ffn_kernel<1024, true><<<grid_s, 256, 0, stream>>>(
        h, wbf[3], wbf[4], wbf[5], nullptr, nullptr, nullptr, out, T);
  } else {
    expert_ffn_kernel<I_RT, false><<<grid_r, 256, 0, stream>>>(
        h, d_in[3], d_in[4], d_in[5], tok_list, wgt_list, counts, out, T);
    expert_ffn_kernel<1024, false><<<grid_s, 256, 0, stream>>>(
        h, d_in[6], d_in[7], d_in[8], nullptr, nullptr, nullptr, out, T);
  }
}
